// MyRNN_35691178229866
// MI455X (gfx1250) — matrix operand order repaired and checked by reading, not compiled, not run on hardware
//
#include <hip/hip_runtime.h>

#define BATCH 512
#define SEQ   256
#define XDIM  512
#define HDIM  1024
#define KTOT  1536   // HDIM + XDIM
#define NKC   24     // KTOT / 64 : total K chunks
#define NXC   16     // chunks 0..15 come from H (bf16), 16..23 from X (fp32)

typedef __attribute__((ext_vector_type(16))) __bf16 v16bf;
typedef __attribute__((ext_vector_type(8)))  float  v8f;

union FragBF { v16bf v; uint4 u[2]; };
union AccF   { v8f   v; float f[8]; };

__device__ __forceinline__ unsigned short f2bf(float x) {
    __bf16 h = (__bf16)x;
    return __builtin_bit_cast(unsigned short, h);
}

// generic pointer -> 32-bit LDS byte offset (AS3 pointers are LDS offsets)
typedef __attribute__((address_space(3))) unsigned char lds_byte_t;
__device__ __forceinline__ unsigned ldsAddr(const void* p) {
    return (unsigned)(unsigned long long)(lds_byte_t*)p;
}

// CDNA5 async DMA: global -> LDS, 16 bytes per lane, tracked by ASYNCcnt.
// GVS addressing: mem = SGPR_base + VGPR_offset ; LDS dest from VDST vgpr.
__device__ __forceinline__ void asyncLoadB128(unsigned lds, unsigned voff,
                                              const void* base) {
    asm volatile("global_load_async_to_lds_b128 %0, %1, %2"
                 :: "v"(lds), "v"(voff), "s"(base)
                 : "memory");
}

// ---------------------------------------------------------------------------
// Prologue: Wcat[n][k] = bf16( k < HDIM ? W_hh[n][k] : W_xh[n][k-HDIM] )
// ---------------------------------------------------------------------------
__global__ __launch_bounds__(256)
void build_wcat(const float* __restrict__ Whh,
                const float* __restrict__ Wxh,
                unsigned short* __restrict__ Wcat) {
    int idx = blockIdx.x * 256 + threadIdx.x;      // 0 .. HDIM*KTOT-1
    int n = idx / KTOT;
    int k = idx - n * KTOT;
    float v = (k < HDIM) ? Whh[n * HDIM + k] : Wxh[n * XDIM + (k - HDIM)];
    Wcat[idx] = f2bf(v);
}

__global__ __launch_bounds__(256)
void zero_u16(unsigned short* __restrict__ p) {
    p[blockIdx.x * 256 + threadIdx.x] = 0;
}

// ---------------------------------------------------------------------------
// One recurrence step, triple-buffered async-DMA pipeline:
//   acc[m][n] = sum_k A[m][k] * Wcat[n][k] + b[n],  A = [Hin_bf16 | bf16(X_t)]
//   if !last : Hout[m][n] = bf16( sigmoid(acc) )   (coalesced via LDS staging)
//   else     : Hlin[m][n] = acc (fp32)
// Block: 128 threads (4 waves), tile 64x64, K-chunks of 64 (2 WMMA k-steps).
// ---------------------------------------------------------------------------
__global__ __launch_bounds__(128, 1)
void rnn_step(const unsigned short* __restrict__ Hin,   // [BATCH][HDIM] bf16
              const float*          __restrict__ X,     // [BATCH][SEQ][XDIM] fp32
              const unsigned short* __restrict__ Wcat,  // [HDIM][KTOT] bf16
              const float*          __restrict__ bias,  // [HDIM]
              unsigned short*       __restrict__ Hout,  // [BATCH][HDIM] bf16
              float*                __restrict__ Hlin,  // [BATCH][HDIM] fp32
              int t, int is_last)
{
    // 3-deep ring; row pad of 8 u16 keeps every row 16B aligned
    __shared__ __align__(16) unsigned short As[3][64][72];
    __shared__ __align__(16) unsigned short Bs[3][64][72];

    const int tid  = threadIdx.x;
    const int lane = tid & 31;
    const int w    = tid >> 5;           // wave 0..3
    const int wm   = (w >> 1) * 32;      // wave M offset within tile
    const int wn   = (w & 1)  * 32;      // wave N offset within tile
    const int m0   = blockIdx.y * 64;
    const int n0   = blockIdx.x * 64;

    AccF c[2][2];
#pragma unroll
    for (int i = 0; i < 2; ++i)
#pragma unroll
        for (int j = 0; j < 2; ++j)
#pragma unroll
            for (int v = 0; v < 8; ++v) c[i][j].f[v] = 0.0f;

    const int kg   = tid & 7;            // 8-element k-group within chunk
    const int lrow = tid >> 3;           // 0..15

    // issue all copies for chunk kc into ring buffer bi
    auto issueChunk = [&](int kc, int bi) {
        const int gkb = kc * 64 + kg * 8;
#pragma unroll
        for (int p = 0; p < 4; ++p) {
            const int r = lrow + p * 16;
            // B tile (Wcat rows): always async DMA
            asyncLoadB128(ldsAddr(&Bs[bi][r][kg * 8]),
                          (unsigned)(((n0 + r) * KTOT + gkb) * 2), Wcat);
            if (gkb < HDIM) {
                // A tile, H region (bf16): async DMA
                asyncLoadB128(ldsAddr(&As[bi][r][kg * 8]),
                              (unsigned)(((m0 + r) * HDIM + gkb) * 2), Hin);
            } else {
                // A tile, X region: fp32 -> bf16 through VGPRs (transient)
                const float* src = &X[(size_t)(m0 + r) * SEQ * XDIM +
                                      (size_t)t * XDIM + (gkb - HDIM)];
                float4 f0 = *(const float4*)src;
                float4 f1 = *(const float4*)(src + 4);
                union { uint4 u; unsigned short h[8]; } cv;
                cv.h[0] = f2bf(f0.x); cv.h[1] = f2bf(f0.y);
                cv.h[2] = f2bf(f0.z); cv.h[3] = f2bf(f0.w);
                cv.h[4] = f2bf(f1.x); cv.h[5] = f2bf(f1.y);
                cv.h[6] = f2bf(f1.z); cv.h[7] = f2bf(f1.w);
                *(uint4*)&As[bi][r][kg * 8] = cv.u;
            }
        }
    };

    auto compute = [&](const unsigned short (*A)[72],
                       const unsigned short (*B)[72]) {
#pragma unroll
        for (int ks = 0; ks < 2; ++ks) {
            const int k0    = ks * 32;
            const int akoff = (lane >> 4) << 3;   // A: lanes 16-31 hold k+8 / k+24
            const int bkoff = (lane >> 4) << 3;   // B: lanes 16-31 hold k+8 / k+24
            FragBF a[2], b[2];
#pragma unroll
            for (int i = 0; i < 2; ++i) {
                const int am = wm + i * 16 + (lane & 15);
                a[i].u[0] = *(const uint4*)&A[am][k0 + akoff];
                a[i].u[1] = *(const uint4*)&A[am][k0 + 16 + akoff];
            }
#pragma unroll
            for (int j = 0; j < 2; ++j) {
                const int bn = wn + j * 16 + (lane & 15);
                b[j].u[0] = *(const uint4*)&B[bn][k0 + bkoff];
                b[j].u[1] = *(const uint4*)&B[bn][k0 + 16 + bkoff];
            }
#pragma unroll
            for (int i = 0; i < 2; ++i)
#pragma unroll
                for (int j = 0; j < 2; ++j)
                    c[i][j].v = __builtin_amdgcn_wmma_f32_16x16x32_bf16(
                        false, a[i].v, false, b[j].v,
                        (short)0, c[i][j].v, false, false);
        }
    };

    // ---- pipelined main loop: DMA runs 2 chunks ahead, 1 barrier/chunk ----
    issueChunk(0, 0);
    issueChunk(1, 1);
#pragma unroll
    for (int kc = 0; kc < NKC; ++kc) {
        // chunk kc must be complete; chunk kc+1 may still be in flight.
        // outstanding async ops of chunk kc+1: 8 (H region), 4 (X region), 0 (none)
        if (kc + 1 >= NKC) {
            asm volatile("s_wait_asynccnt 0x0" ::: "memory");
        } else if (kc + 1 < NXC) {
            asm volatile("s_wait_asynccnt 0x8" ::: "memory");
        } else {
            asm volatile("s_wait_asynccnt 0x4" ::: "memory");
        }
        __syncthreads();                         // chunk kc visible to all waves
        if (kc + 2 < NKC) issueChunk(kc + 2, (kc + 2) % 3);
        compute(As[kc % 3], Bs[kc % 3]);
    }

    // ---- epilogue: bias + activation, staged through LDS for coalescing ----
    const int n_l  = lane & 15;
    const int mrow = (lane >> 4) << 3;   // lanes 16-31 hold rows m+8..m+15
    __syncthreads();                     // all compute done; LDS reusable

    if (!is_last) {
        unsigned short (*Ss)[72] = As[0];            // 64 x 72 u16 staging
#pragma unroll
        for (int i = 0; i < 2; ++i) {
#pragma unroll
            for (int j = 0; j < 2; ++j) {
                const int col = wn + j * 16 + n_l;
                const float bb = bias[n0 + col];
#pragma unroll
                for (int v = 0; v < 8; ++v) {
                    const int row = wm + i * 16 + mrow + v;
                    const float val = c[i][j].f[v] + bb;
                    Ss[row][col] = f2bf(1.0f / (1.0f + __expf(-val)));
                }
            }
        }
        __syncthreads();
        const int r  = tid >> 1;                     // 0..63
        const int c0 = (tid & 1) * 32;               // 0 or 32
        unsigned short* dst = &Hout[(size_t)(m0 + r) * HDIM + n0 + c0];
#pragma unroll
        for (int q = 0; q < 4; ++q)
            *(uint4*)(dst + q * 8) = *(const uint4*)&Ss[r][c0 + q * 8];
    } else {
        float (*Sf)[68] = (float (*)[68]) & As[0][0][0];   // 64 x 68 f32 (17.0 KB)
#pragma unroll
        for (int i = 0; i < 2; ++i) {
#pragma unroll
            for (int j = 0; j < 2; ++j) {
                const int col = wn + j * 16 + n_l;
                const float bb = bias[n0 + col];
#pragma unroll
                for (int v = 0; v < 8; ++v) {
                    const int row = wm + i * 16 + mrow + v;
                    Sf[row][col] = c[i][j].f[v] + bb;
                }
            }
        }
        __syncthreads();
        const int r  = tid >> 1;
        const int c0 = (tid & 1) * 32;
        float* dst = &Hlin[(size_t)(m0 + r) * HDIM + n0 + c0];
#pragma unroll
        for (int q = 0; q < 8; ++q)
            *(uint4*)(dst + q * 4) = *(const uint4*)&Sf[r][c0 + q * 4];
    }
}

// ---------------------------------------------------------------------------
// Epilogue: out[b] = sum_h Hlin[b][h] * W_hy[h]
// ---------------------------------------------------------------------------
__global__ __launch_bounds__(256)
void final_dot(const float* __restrict__ Hlin,
               const float* __restrict__ Why,
               float* __restrict__ out) {
    __shared__ float red[256];
    const int b = blockIdx.x;
    float s = 0.0f;
    for (int h = threadIdx.x; h < HDIM; h += 256)
        s += Hlin[(size_t)b * HDIM + h] * Why[h];
    red[threadIdx.x] = s;
    __syncthreads();
    for (int off = 128; off > 0; off >>= 1) {
        if (threadIdx.x < off) red[threadIdx.x] += red[threadIdx.x + off];
        __syncthreads();
    }
    if (threadIdx.x == 0) out[b] = red[0];
}

// ---------------------------------------------------------------------------
extern "C" void kernel_launch(void* const* d_in, const int* in_sizes, int n_in,
                              void* d_out, int out_size, void* d_ws, size_t ws_size,
                              hipStream_t stream) {
    (void)in_sizes; (void)n_in; (void)out_size; (void)ws_size;
    const float* X    = (const float*)d_in[0];   // [B,S,X]
    const float* Whh  = (const float*)d_in[1];   // [H,H]
    const float* Wxh  = (const float*)d_in[2];   // [H,X]
    const float* Why  = (const float*)d_in[3];   // [H]
    const float* bias = (const float*)d_in[4];   // [H]
    float* out = (float*)d_out;                  // [B]

    char* ws = (char*)d_ws;
    unsigned short* Wcat = (unsigned short*)ws;                       // 3,145,728 B
    unsigned short* H0   = (unsigned short*)(ws + 3145728);           // 1,048,576 B
    unsigned short* H1   = (unsigned short*)(ws + 3145728 + 1048576); // 1,048,576 B
    float*          Hlin = (float*)(ws + 3145728 + 2 * 1048576);      // 2,097,152 B

    build_wcat<<<(HDIM * KTOT) / 256, 256, 0, stream>>>(Whh, Wxh, Wcat);
    zero_u16<<<(BATCH * HDIM) / 256, 256, 0, stream>>>(H0);

    unsigned short* bufs[2] = {H0, H1};
    dim3 grid(HDIM / 64, BATCH / 64);   // 16 x 8 = 128 workgroups / step
    for (int t = 0; t < SEQ; ++t) {
        const int last = (t == SEQ - 1);
        rnn_step<<<grid, 128, 0, stream>>>(bufs[t & 1], X, Wcat, bias,
                                           bufs[(t + 1) & 1], Hlin, t, last);
    }
    final_dot<<<BATCH, 256, 0, stream>>>(Hlin, Why, out);
}
